// GNN_85487029060183
// MI455X (gfx1250) — matrix operand order repaired and checked by reading, not compiled, not run on hardware
//
#include <hip/hip_runtime.h>

// Problem constants (fixed by the reference)
#define NNODES 32768
#define DEG    16
#define CH     128
#define G4     512   // 4*CH gate width

typedef _Float16    half16 __attribute__((ext_vector_type(16)));
typedef float       f32x8  __attribute__((ext_vector_type(8)));
typedef unsigned int u32x4 __attribute__((ext_vector_type(4)));

union ABFrag { half16 h; u32x4 q[2]; };

// Packed B-fragment geometry: per fragment (one 32x16 f16 B tile):
//   32 lanes x 32 contiguous bytes, +16B pad every 8 lanes  -> 1088 B
// (pad keeps ds_load_b128 at worst 2-way bank conflicted).
#define FRAGB   1088
// LSTM LDS stage: 16 rows x 512B data, row stride padded to 544B (8-bank skew)
#define STG_ROW   544
#define STG_WAVE  8704            // 16 * 544
#define STG_BUF   69632           // 8 waves * STG_WAVE
// h state: 32 rows x 256B data, row stride padded to 272B
#define H_ROW     272
#define LDS_W     139264          // 128 fragments * FRAGB
#define LDS_TOTAL (LDS_W + 2*STG_BUF + 32*H_ROW)   // 287232 B

static __device__ __forceinline__ float sigmoidf_(float x) {
  return 1.0f / (1.0f + __expf(-x));
}

static __device__ __forceinline__ const char* frag_addr(const char* base,
                                                        int frag, int lane) {
  return base + (size_t)frag * FRAGB + lane * 32 + (lane >> 3) * 16;
}

static __device__ __forceinline__ void load_frag(ABFrag& f, const char* p) {
  f.q[0] = *(const u32x4*)p;
  f.q[1] = *(const u32x4*)(p + 16);
}

// ---------------------------------------------------------------------------
// Prep kernels
// ---------------------------------------------------------------------------

__global__ void k_f32_to_f16(const float* __restrict__ in,
                             _Float16* __restrict__ out, int n) {
  int i = blockIdx.x * 256 + threadIdx.x;
  if (i < n) out[i] = (_Float16)in[i];
}

// Pack W[O x C] (row-major f32) into padded B-fragment blobs for
// v_wmma_f32_16x16x32_f16: B element (k, n) = W[n][k].
// lane%16 <-> n; 16 consecutive halves per lane <-> k in table order (h = lane/16).
__global__ void k_pack_b(const float* __restrict__ W,
                         char* __restrict__ out, int O, int C) {
  int id = blockIdx.x * 256 + threadIdx.x;
  int total = (O / 16) * (C / 32) * 32;
  if (id >= total) return;
  int lane = id & 31;
  int frag = id >> 5;                 // = ntile*(C/32) + kb
  int kb    = frag % (C / 32);
  int ntile = frag / (C / 32);
  int n = ntile * 16 + (lane & 15);
  int hb = (lane >> 4) * 8;
  _Float16* dst = (_Float16*)frag_addr(out, frag, lane);
#pragma unroll
  for (int j = 0; j < 16; j++)
    dst[j] = (_Float16)W[(size_t)n * C + kb * 32 + (j < 8 ? hb + j : 8 + hb + j)];
}

__global__ void k_bias_sum(const float* __restrict__ a,
                           const float* __restrict__ b,
                           float* __restrict__ out, int n) {
  int i = blockIdx.x * 256 + threadIdx.x;
  if (i < n) out[i] = a[i] + b[i];
}

// ---------------------------------------------------------------------------
// U = X16 @ Wih^T + (bih+bhh)   -> [N, 512] f32
// One wave computes a 16-row x 128-col tile of U.  B fragments are loaded as
// double-buffered groups of 4 (8 b128 loads in flight per wait point).
// ---------------------------------------------------------------------------
__global__ __launch_bounds__(256) void k_input_gemm(
    const _Float16* __restrict__ X16,   // [N, CH] f16 row-major
    const char* __restrict__ WP,        // padded packed fragments
    const float* __restrict__ bias,     // [G4]
    float* __restrict__ U)              // [N, G4]
{
  const int lane    = threadIdx.x & 31;
  const int gwave   = blockIdx.x * 8 + (threadIdx.x >> 5);
  const int rowTile = gwave >> 2;
  const int chunk   = gwave & 3;
  const int mA      = lane & 15;
  const int hiSel   = lane >> 4;

  // A fragments from a row-major f16 row (16-bit A 16x32 layout)
  ABFrag a[4];
  const char* xrow = (const char*)(X16 + (size_t)(rowTile * 16 + mA) * CH);
#pragma unroll
  for (int kb = 0; kb < 4; kb++) {
    const char* p = xrow + kb * 64 + hiSel * 16;
    a[kb].q[0] = *(const u32x4*)p;
    a[kb].q[1] = *(const u32x4*)(p + 32);
  }

  f32x8 acc[8];
#pragma unroll
  for (int j = 0; j < 8; j++) {
    float bv = bias[chunk * 128 + j * 16 + mA];
#pragma unroll
    for (int r = 0; r < 8; r++) acc[j][r] = bv;
  }

  // 32 WMMAs in 8 groups of 4: group g -> kb = g>>1, j = (g&1)*4 ..
  ABFrag b[2][4];
#pragma unroll
  for (int j = 0; j < 4; j++)
    load_frag(b[0][j], frag_addr(WP, (chunk * 8 + j) * 4 + 0, lane));
#pragma unroll
  for (int gi = 0; gi < 8; gi++) {
    if (gi + 1 < 8) {
      int gn = gi + 1, kbn = gn >> 1, jb = (gn & 1) * 4;
#pragma unroll
      for (int j = 0; j < 4; j++)
        load_frag(b[gn & 1][j],
                  frag_addr(WP, (chunk * 8 + jb + j) * 4 + kbn, lane));
    }
    int kb = gi >> 1, jb = (gi & 1) * 4;
#pragma unroll
    for (int j = 0; j < 4; j++)
      acc[jb + j] = __builtin_amdgcn_wmma_f32_16x16x32_f16(
          false, a[kb].h, false, b[gi & 1][j].h, (short)0,
          acc[jb + j], false, false);
  }

#pragma unroll
  for (int j = 0; j < 8; j++) {
#pragma unroll
    for (int r = 0; r < 8; r++) {
      int m = r + hiSel * 8;
      U[(size_t)(rowTile * 16 + m) * G4 + chunk * 128 + j * 16 + mA] = acc[j][r];
    }
  }
}

// ---------------------------------------------------------------------------
// LSTM aggregation over D=16 neighbors.
// Workgroup: 8 waves, 32 nodes.  wave = (node-tile nt, gate-chunk).
// LDS: [0, 139264)        packed Whh^T fragments (padded)
//      [139264, +2*69632) double-buffered U-stage / raw-gate scratch
//      [278528, +8704)    h state, f16, padded rows
// Gather for t+1 is issued into registers before the WMMAs of step t; the
// register block is committed to the other LDS stage buffer after the WMMAs.
// B fragments double-buffered in groups of 2 (keeps peak VGPRs ~210).
// ---------------------------------------------------------------------------
__global__ __launch_bounds__(256) void k_lstm(
    const float* __restrict__ U,        // [N, G4]
    const int*  __restrict__ src,       // [N*DEG]
    const char* __restrict__ whhP,      // packed fragments, 139264 B
    _Float16* __restrict__ Hout)        // [N, CH] final hidden (aggr), f16
{
  extern __shared__ char smem[];
  char* sW = smem;                       // packed Whh fragments
  char* sStage0 = smem + LDS_W;          // stage/gates buffer 0
  char* sStage1 = sStage0 + STG_BUF;     // stage/gates buffer 1
  char* sH = sStage1 + STG_BUF;          // h state

  const int tid   = threadIdx.x;
  const int lane  = tid & 31;
  const int wave  = tid >> 5;     // 0..7
  const int nt    = wave >> 2;    // node tile 0/1
  const int chunk = wave & 3;     // gate chunk (i,f,g,o)
  const int nodeBase = blockIdx.x * 32;
  const int mA    = lane & 15;
  const int hiSel = lane >> 4;

  // stage packed Whh fragments into LDS (b128 copies)
  {
    const u32x4* g = (const u32x4*)whhP;
    u32x4* d = (u32x4*)sW;
    for (int i = tid; i < LDS_W / 16; i += 256) d[i] = g[i];
  }
  // zero h state (including pad)
  {
    u32x4 z = {0u, 0u, 0u, 0u};
    u32x4* d = (u32x4*)sH;
    for (int i = tid; i < (32 * H_ROW) / 16; i += 256) d[i] = z;
  }

  float c[16];
#pragma unroll
  for (int e = 0; e < 16; e++) c[e] = 0.f;

  const int gRow  = lane >> 1;          // row 0..15 handled by a lane pair
  const int gSub  = (lane & 1) * 16;    // 16B sub-offset within 32B strip
  const int gnode = nodeBase + nt * 16 + gRow;

  // ---- preload t=0 gather into stage buffer 0 ----
  u32x4 g[16];
  {
    int s = src[gnode * DEG + 0];
    const char* urow = (const char*)(U + (size_t)s * G4 + chunk * 128) + gSub;
#pragma unroll
    for (int i = 0; i < 16; i++) g[i] = *(const u32x4*)(urow + i * 32);
    char* stg = sStage0 + wave * STG_WAVE + gRow * STG_ROW + gSub;
#pragma unroll
    for (int i = 0; i < 16; i++) *(u32x4*)(stg + i * 32) = g[i];
  }
  __syncthreads();

  for (int t = 0; t < DEG; t++) {
    char* bufT = (t & 1) ? sStage1 : sStage0;
    char* bufN = (t & 1) ? sStage0 : sStage1;

    // ---- issue global gather for t+1 into registers (latency hidden) ----
    if (t + 1 < DEG) {
      int s = src[gnode * DEG + t + 1];
      const char* urow = (const char*)(U + (size_t)s * G4 + chunk * 128) + gSub;
#pragma unroll
      for (int i = 0; i < 16; i++) g[i] = *(const u32x4*)(urow + i * 32);
      if (t + 2 < DEG) {
        int s2 = src[gnode * DEG + t + 2];
        __builtin_prefetch((const void*)(U + (size_t)s2 * G4 + chunk * 128), 0, 3);
      }
    }

    // ---- h A-fragments from LDS (padded rows) ----
    ABFrag a[4];
    {
      const char* hrow = sH + (nt * 16 + mA) * H_ROW;
#pragma unroll
      for (int kb = 0; kb < 4; kb++) {
        const char* p = hrow + kb * 64 + hiSel * 16;
        a[kb].q[0] = *(const u32x4*)p;
        a[kb].q[1] = *(const u32x4*)(p + 32);
      }
    }

    // ---- accumulator init from staged U (D-layout, padded stride) ----
    f32x8 acc[8];
    {
      const float* stgf = (const float*)(bufT + wave * STG_WAVE);
#pragma unroll
      for (int j = 0; j < 8; j++)
#pragma unroll
        for (int r = 0; r < 8; r++)
          acc[j][r] = stgf[(r + hiSel * 8) * (STG_ROW / 4) + j * 16 + mA];
    }

    // ---- gates += h @ Whh^T : 32 WMMAs, 16 groups of 2 LDS B frags ----
    {
      ABFrag b[2][2];
#pragma unroll
      for (int j = 0; j < 2; j++)
        load_frag(b[0][j], frag_addr(sW, (chunk * 8 + j) * 4 + 0, lane));
#pragma unroll
      for (int gi = 0; gi < 16; gi++) {
        if (gi + 1 < 16) {
          int gn = gi + 1, kbn = gn >> 2, jb = (gn & 3) * 2;
#pragma unroll
          for (int j = 0; j < 2; j++)
            load_frag(b[gn & 1][j],
                      frag_addr(sW, (chunk * 8 + jb + j) * 4 + kbn, lane));
        }
        int kb = gi >> 2, jb = (gi & 3) * 2;
#pragma unroll
        for (int j = 0; j < 2; j++)
          acc[jb + j] = __builtin_amdgcn_wmma_f32_16x16x32_f16(
              false, a[kb].h, false, b[gi & 1][j].h, (short)0,
              acc[jb + j], false, false);
      }
    }

    // ---- store raw gates into own region (aliases consumed stage) ----
    {
      float* gts = (float*)(bufT + wave * STG_WAVE);
#pragma unroll
      for (int j = 0; j < 8; j++)
#pragma unroll
        for (int r = 0; r < 8; r++)
          gts[(r + hiSel * 8) * (STG_ROW / 4) + j * 16 + mA] = acc[j][r];
    }

    // ---- commit t+1 gather registers into the other stage buffer ----
    if (t + 1 < DEG) {
      char* stg = bufN + wave * STG_WAVE + gRow * STG_ROW + gSub;
#pragma unroll
      for (int i = 0; i < 16; i++) *(u32x4*)(stg + i * 32) = g[i];
    }
    __syncthreads();

    // ---- elementwise LSTM cell update (gate order i,f,g,o = chunks 0..3) ----
#pragma unroll
    for (int e = 0; e < 16; e++) {
      int flat = tid + e * 256;              // node*128 + col, node in [0,32)
      int node = flat >> 7, col = flat & 127;
      const char* waveBase = bufT + ((node >> 4) * 4) * STG_WAVE
                                  + (node & 15) * STG_ROW + col * 4;
      float gi = *(const float*)(waveBase);
      float gf = *(const float*)(waveBase + STG_WAVE);
      float gg = *(const float*)(waveBase + 2 * STG_WAVE);
      float go = *(const float*)(waveBase + 3 * STG_WAVE);
      float si = sigmoidf_(gi), sf = sigmoidf_(gf), so = sigmoidf_(go);
      c[e] = sf * c[e] + si * tanhf(gg);
      ((_Float16*)sH)[node * (H_ROW / 2) + col] =
          (_Float16)(so * tanhf(c[e]));
    }
    __syncthreads();
  }

  // write final hidden state (the LSTM aggregation) as f16 row-major
  {
    u32x4* d = (u32x4*)(Hout + (size_t)nodeBase * CH);
    for (int i = tid; i < 512; i += 256) {
      int row = i >> 4, sub = i & 15;
      d[i] = *(const u32x4*)(sH + row * H_ROW + sub * 16);
    }
  }
}

// ---------------------------------------------------------------------------
// out = aggr @ Wl^T + bl + x @ Wr^T   (+ optional ReLU, f16 or f32 store)
// One wave computes a 16x128 output tile.  64 WMMAs in 16 double-buffered
// groups of 4 spanning both weight matrices.
// ---------------------------------------------------------------------------
__global__ __launch_bounds__(256) void k_out_gemm(
    const _Float16* __restrict__ Agg16,   // [N, CH] f16
    const _Float16* __restrict__ Xin16,   // [N, CH] f16
    const char* __restrict__ WlP,         // packed fragments
    const char* __restrict__ WrP,         // packed fragments
    const float* __restrict__ bl,         // [CH]
    float* __restrict__ outF,             // f32 out (mode 0)
    _Float16* __restrict__ outH,          // f16 out (mode 1, with ReLU)
    int storeHalfRelu)
{
  const int lane    = threadIdx.x & 31;
  const int rowTile = blockIdx.x * 8 + (threadIdx.x >> 5);
  const int mA      = lane & 15;
  const int hiSel   = lane >> 4;

  f32x8 acc[8];
#pragma unroll
  for (int j = 0; j < 8; j++) {
    float bv = bl[j * 16 + mA];
#pragma unroll
    for (int r = 0; r < 8; r++) acc[j][r] = bv;
  }

  const _Float16* mats[2] = {Agg16, Xin16};
  const char*     wps[2]  = {WlP, WrP};

  // preload both A fragment sets
  ABFrag a[2][4];
#pragma unroll
  for (int m2 = 0; m2 < 2; m2++) {
    const char* arow = (const char*)(mats[m2] + (size_t)(rowTile * 16 + mA) * CH);
#pragma unroll
    for (int kb = 0; kb < 4; kb++) {
      const char* p = arow + kb * 64 + hiSel * 16;
      a[m2][kb].q[0] = *(const u32x4*)p;
      a[m2][kb].q[1] = *(const u32x4*)(p + 32);
    }
  }

  // 64 WMMAs in 16 groups of 4: group g -> m2 = g>>3, kb = (g>>1)&3,
  // j = (g&1)*4 ..  (B double-buffered across the matrix boundary)
  ABFrag b[2][4];
#pragma unroll
  for (int j = 0; j < 4; j++)
    load_frag(b[0][j], frag_addr(wps[0], j * 4 + 0, lane));
#pragma unroll
  for (int gi = 0; gi < 16; gi++) {
    if (gi + 1 < 16) {
      int gn = gi + 1, m2n = gn >> 3, kbn = (gn >> 1) & 3, jb = (gn & 1) * 4;
#pragma unroll
      for (int j = 0; j < 4; j++)
        load_frag(b[gn & 1][j],
                  frag_addr(wps[m2n], (jb + j) * 4 + kbn, lane));
    }
    int m2 = gi >> 3, kb = (gi >> 1) & 3, jb = (gi & 1) * 4;
#pragma unroll
    for (int j = 0; j < 4; j++)
      acc[jb + j] = __builtin_amdgcn_wmma_f32_16x16x32_f16(
          false, a[m2][kb].h, false, b[gi & 1][j].h, (short)0,
          acc[jb + j], false, false);
  }

#pragma unroll
  for (int j = 0; j < 8; j++) {
#pragma unroll
    for (int r = 0; r < 8; r++) {
      int m = r + hiSel * 8;
      size_t idx = (size_t)(rowTile * 16 + m) * CH + j * 16 + mA;
      float v = acc[j][r];
      if (storeHalfRelu) {
        v = v > 0.f ? v : 0.f;
        outH[idx] = (_Float16)v;
      } else {
        outF[idx] = v;
      }
    }
  }
}

// ---------------------------------------------------------------------------
// Host launcher
// ---------------------------------------------------------------------------
extern "C" void kernel_launch(void* const* d_in, const int* in_sizes, int n_in,
                              void* d_out, int out_size, void* d_ws, size_t ws_size,
                              hipStream_t stream) {
  (void)in_sizes; (void)n_in; (void)out_size; (void)ws_size;

  const float* x    = (const float*)d_in[0];
  const int*   edge = (const int*)d_in[1];   // [2, E]; first E entries = src
  const float* w1ih = (const float*)d_in[2];
  const float* w1hh = (const float*)d_in[3];
  const float* b1ih = (const float*)d_in[4];
  const float* b1hh = (const float*)d_in[5];
  const float* w1l  = (const float*)d_in[6];
  const float* b1l  = (const float*)d_in[7];
  const float* w1r  = (const float*)d_in[8];
  const float* w2ih = (const float*)d_in[9];
  const float* w2hh = (const float*)d_in[10];
  const float* b2ih = (const float*)d_in[11];
  const float* b2hh = (const float*)d_in[12];
  const float* w2l  = (const float*)d_in[13];
  const float* b2l  = (const float*)d_in[14];
  const float* w2r  = (const float*)d_in[15];

  char* ws = (char*)d_ws;
  float*    U  = (float*)ws;                               // 64 MB
  _Float16* Xa = (_Float16*)(ws + ((size_t)64 << 20));     // 8 MB layer-1 input f16
  _Float16* Xb = (_Float16*)(ws + ((size_t)72 << 20));     // 8 MB layer-2 input f16
  _Float16* Hh = (_Float16*)(ws + ((size_t)80 << 20));     // 8 MB aggr f16
  char*     P  = ws + ((size_t)88 << 20);                  // packed weights
  const size_t PACK_IH = 128 * FRAGB;   // 139264 B (Wih / Whh: 128 fragments)
  const size_t PACK_L  = 32 * FRAGB;    //  34816 B (Wl / Wr: 32 fragments)
  char* w1ihP = P;
  char* w1hhP = w1ihP + PACK_IH;
  char* w1lP  = w1hhP + PACK_IH;
  char* w1rP  = w1lP + PACK_L;
  char* w2ihP = w1rP + PACK_L;
  char* w2hhP = w2ihP + PACK_IH;
  char* w2lP  = w2hhP + PACK_IH;
  char* w2rP  = w2lP + PACK_L;
  float* b1 = (float*)(w2rP + PACK_L);
  float* b2 = b1 + G4;

  // -------- prep: f16 convert + weight packing + fused biases --------
  k_f32_to_f16<<<(NNODES * CH + 255) / 256, 256, 0, stream>>>(x, Xa, NNODES * CH);
  k_pack_b<<<16, 256, 0, stream>>>(w1ih, w1ihP, G4, CH);
  k_pack_b<<<16, 256, 0, stream>>>(w1hh, w1hhP, G4, CH);
  k_pack_b<<<4, 256, 0, stream>>>(w1l, w1lP, CH, CH);
  k_pack_b<<<4, 256, 0, stream>>>(w1r, w1rP, CH, CH);
  k_pack_b<<<16, 256, 0, stream>>>(w2ih, w2ihP, G4, CH);
  k_pack_b<<<16, 256, 0, stream>>>(w2hh, w2hhP, G4, CH);
  k_pack_b<<<4, 256, 0, stream>>>(w2l, w2lP, CH, CH);
  k_pack_b<<<4, 256, 0, stream>>>(w2r, w2rP, CH, CH);
  k_bias_sum<<<2, 256, 0, stream>>>(b1ih, b1hh, b1, G4);
  k_bias_sum<<<2, 256, 0, stream>>>(b2ih, b2hh, b2, G4);

  // -------- layer 1 --------
  k_input_gemm<<<1024, 256, 0, stream>>>(Xa, w1ihP, b1, U);
  k_lstm<<<1024, 256, LDS_TOTAL, stream>>>(U, edge, w1hhP, Hh);
  k_out_gemm<<<256, 256, 0, stream>>>(Hh, Xa, w1lP, w1rP, b1l,
                                      nullptr, Xb, 1);

  // -------- layer 2 --------
  k_input_gemm<<<1024, 256, 0, stream>>>(Xb, w2ihP, b2, U);
  k_lstm<<<1024, 256, LDS_TOTAL, stream>>>(U, edge, w2hhP, Hh);
  k_out_gemm<<<256, 256, 0, stream>>>(Hh, Xb, w2lP, w2rP, b2l,
                                      (float*)d_out, nullptr, 0);
}
